// SimpleAttention_70703751627380
// MI455X (gfx1250) — matrix operand order repaired and checked by reading, not compiled, not run on hardware
//
#include <hip/hip_runtime.h>

// ---------------------------------------------------------------------------
// SimpleAttention for MI455X (gfx1250, wave32, WMMA + Tensor Data Mover).
// Pipeline: f32->f16 convert -> QKV GEMM (TDM-fed WMMA, double-buffered LDS)
// -> causal flash attention (WMMA; TDM for K tiles) -> output GEMM.
// ---------------------------------------------------------------------------

typedef _Float16 v16h __attribute__((ext_vector_type(16)));
typedef _Float16 v8h  __attribute__((ext_vector_type(8)));
typedef float    v8f  __attribute__((ext_vector_type(8)));

// ---- problem constants (match reference) ----------------------------------
#define BATCH     4
#define SEQ       2048
#define HIDDEN    2048
#define NUM_HEADS 16
#define HEAD_DIM  128          // HIDDEN / NUM_HEADS
#define H3        (3 * HIDDEN) // 6144
#define MROWS     (BATCH * SEQ)        // 8192

// ---------------------------------------------------------------------------
// Tensor Data Mover support. This toolchain exposes the 6-arg builtin:
//   (uint32x4 g0, int32x8 g1, int32x4 g2, int32x4 g3, int32x8 extra, i32 cpol)
// Groups 2/3 (and the extra group) are unused for <=2D tensors -> zero-filled.
// ---------------------------------------------------------------------------
#if defined(__gfx1250__) && __has_builtin(__builtin_amdgcn_tensor_load_to_lds)
#define HAVE_TDM 1
typedef unsigned int u32x4 __attribute__((ext_vector_type(4)));
typedef int          i32x8 __attribute__((ext_vector_type(8)));
typedef int          i32x4 __attribute__((ext_vector_type(4)));
typedef __attribute__((address_space(3))) const void* lds_cptr_t;

__device__ __forceinline__ unsigned lds_addr_of(const void* p) {
  // generic -> LDS addrspace -> 32-bit workgroup-relative byte offset
  return (unsigned)(unsigned long long)(lds_cptr_t)p;
}

// 2D f16 tile DMA, global (row-major) -> LDS, with per-row LDS padding.
// tile_cols f16 per row, tile_rows rows, global row stride in elements.
// pad: insert (amt_code+1) DWORDs every 2^(ivl_code+1) DWORDs stored.
__device__ __forceinline__ void tdm_load_tile_f16(
    const void* lds_dst, const _Float16* gsrc,
    int tile_cols, int tile_rows, int row_stride_elems,
    int ivl_code, int amt_code) {
  const unsigned long long ga = (unsigned long long)gsrc;
  u32x4 g0;
  g0[0] = 1u;                                   // count=1 (valid user D#)
  g0[1] = lds_addr_of(lds_dst);                 // lds_addr (bytes)
  g0[2] = (unsigned)ga;                         // global_addr[31:0]
  g0[3] = (unsigned)((ga >> 32) & 0x01FFFFFFu)  // global_addr[56:32]
        | (2u << 30);                           // D#.type = 2
  i32x8 g1;
  g1[0] = (1 << 16)                             // data_size = 2 bytes
        | (1 << 20)                             // pad_enable
        | (ivl_code << 22)                      // pad_interval
        | (amt_code << 25);                     // pad_amount
  g1[1] = (tile_cols & 0xFFFF) << 16;           // tensor_dim0[15:0]
  g1[2] = ((tile_cols >> 16) & 0xFFFF)          // tensor_dim0[31:16]
        | ((tile_rows & 0xFFFF) << 16);         // tensor_dim1[15:0]
  g1[3] = ((tile_rows >> 16) & 0xFFFF)          // tensor_dim1[31:16]
        | ((tile_cols & 0xFFFF) << 16);         // tile_dim0
  g1[4] = (tile_rows & 0xFFFF);                 // tile_dim1 (tile_dim2=0)
  g1[5] = row_stride_elems;                     // tensor_dim0_stride[31:0]
  g1[6] = 0;
  g1[7] = 0;
  const i32x4 z4 = {0, 0, 0, 0};                // groups 2/3 unused (2D)
  const i32x8 z8 = {0, 0, 0, 0, 0, 0, 0, 0};    // extra group unused
  __builtin_amdgcn_tensor_load_to_lds(g0, g1, z4, z4, z8, 0);
}
#define TDM_WAIT0() __builtin_amdgcn_s_wait_tensorcnt(0)
#else
#define HAVE_TDM 0
#endif

// ---------------------------------------------------------------------------
// WMMA helpers: fragment layouts per CDNA5 ISA 7.12.2 (wave32).
// A (16x32 f16): lane l -> row (l&15); K = kb..kb+7 and kb+16..kb+23,
//                kb = (l>>4)*8  -> two contiguous 8xf16 (16B) chunks.
// B (32x16 f16): lane l -> col (l&15); same K order as A (kb..kb+7, kb+16..kb+23).
// C/D (16x16 f32): lane l, vgpr r -> row (l>>4)*8+r, col (l&15).
// ---------------------------------------------------------------------------
__device__ __forceinline__ v16h pack16(v8h lo, v8h hi) {
  return __builtin_shufflevector(lo, hi, 0, 1, 2, 3, 4, 5, 6, 7,
                                 8, 9, 10, 11, 12, 13, 14, 15);
}

__device__ __forceinline__ v16h load_frag_a(const _Float16* p, int ld) {
  const int lane = threadIdx.x & 31;
  const _Float16* q = p + (lane & 15) * ld + ((lane >> 4) << 3);
  v8h lo = *(const v8h*)(q);
  v8h hi = *(const v8h*)(q + 16);
  return pack16(lo, hi);
}

__device__ __forceinline__ v16h load_frag_b(const _Float16* p, int ld) {
  const int lane = threadIdx.x & 31;
  const _Float16* q = p + (lane & 15) * ld + ((lane >> 4) << 3);
  v8h lo = *(const v8h*)(q);
  v8h hi = *(const v8h*)(q + 16);
  return pack16(lo, hi);
}

__device__ __forceinline__ v8f wmma_f16(v16h a, v16h b, v8f c) {
  return __builtin_amdgcn_wmma_f32_16x16x32_f16(false, a, false, b,
                                                (short)0, c, false, false);
}

// ---------------------------------------------------------------------------
// Kernel 1: f32 -> f16 conversion, 8 elements/thread (n % 8 == 0 everywhere)
// ---------------------------------------------------------------------------
__global__ __launch_bounds__(256) void cvt_f32_to_f16(
    const float* __restrict__ in, _Float16* __restrict__ out, long n) {
  long i = ((long)blockIdx.x * 256 + threadIdx.x) * 8;
  if (i + 8 > n) return;
  float4 a = *(const float4*)(in + i);
  float4 b = *(const float4*)(in + i + 4);
  v8h o;
  o[0] = (_Float16)a.x; o[1] = (_Float16)a.y;
  o[2] = (_Float16)a.z; o[3] = (_Float16)a.w;
  o[4] = (_Float16)b.x; o[5] = (_Float16)b.y;
  o[6] = (_Float16)b.z; o[7] = (_Float16)b.w;
  *(v8h*)(out + i) = o;
}

// ---------------------------------------------------------------------------
// Fallback cooperative staging (if no TDM builtin): 256 threads, rows x 32 f16
// ---------------------------------------------------------------------------
__device__ __forceinline__ void stage_tile_manual(
    const _Float16* __restrict__ g, _Float16* lds, int rows, int ldg, int k0) {
  const int t = threadIdx.x;
  for (int rr = 0; rr < rows; rr += 128) {
    const int row = rr + (t >> 1);
    const int col = (t & 1) * 16;
    v8h x0 = *(const v8h*)(g + (size_t)row * ldg + k0 + col);
    v8h x1 = *(const v8h*)(g + (size_t)row * ldg + k0 + col + 8);
    *(v8h*)(lds + row * 40 + col)     = x0;
    *(v8h*)(lds + row * 40 + col + 8) = x1;
  }
}

// ---------------------------------------------------------------------------
// Kernel 2/4: GEMM  C[M,N] = A[M,K] * B[N,K]^T   (f16 in, f16 or f32 out)
// Block tile 128x256, BK=32, double-buffered LDS fed by TDM.
// 8 waves in 2x4; each wave owns 64x64 (4x4 WMMA accs) -> 16 WMMAs/k-step.
// ---------------------------------------------------------------------------
template <bool OUT_F32>
__global__ __launch_bounds__(256) void gemm_wmma_kernel(
    const _Float16* __restrict__ A, const _Float16* __restrict__ B,
    void* __restrict__ Cout, int M, int N, int K) {
  constexpr int BM = 128, BN = 256, BK = 32;
  constexpr int LD = BK + 8;  // 40 f16 = 80B padded row (16B aligned)
  __shared__ _Float16 As[2][BM * LD];
  __shared__ _Float16 Bs[2][BN * LD];

  const int tid  = threadIdx.x;
  const int wave = tid >> 5;
  const int lane = tid & 31;
  const int bm = blockIdx.y * BM;
  const int bn = blockIdx.x * BN;
  const int wm = (wave >> 2) * 64;  // 0 or 64
  const int wn = (wave & 3) * 64;   // 0..192

  const _Float16* Abase = A + (size_t)bm * K;
  const _Float16* Bbase = B + (size_t)bn * K;

  v8f acc[4][4] = {};

  // ---- prologue: fill buffer 0 --------------------------------------------
#if HAVE_TDM
  if (wave == 0)      tdm_load_tile_f16(As[0], Abase, BK, BM, K, 3, 3);
  else if (wave == 1) tdm_load_tile_f16(Bs[0], Bbase, BK, BN, K, 3, 3);
  if (wave < 2) TDM_WAIT0();
#else
  stage_tile_manual(Abase, As[0], BM, K, 0);
  stage_tile_manual(Bbase, Bs[0], BN, K, 0);
#endif
  __syncthreads();

  int buf = 0;
  for (int k0 = 0; k0 < K; k0 += BK) {
    const bool has_next = (k0 + BK) < K;
#if HAVE_TDM
    if (has_next) {  // async DMA into the other buffer, overlapped w/ compute
      if (wave == 0)
        tdm_load_tile_f16(As[buf ^ 1], Abase + k0 + BK, BK, BM, K, 3, 3);
      else if (wave == 1)
        tdm_load_tile_f16(Bs[buf ^ 1], Bbase + k0 + BK, BK, BN, K, 3, 3);
    }
#endif
    // ---- 16 WMMAs from current buffer -------------------------------------
    v16h af[4], bf[4];
#pragma unroll
    for (int i = 0; i < 4; i++)
      af[i] = load_frag_a(&As[buf][(wm + i * 16) * LD], LD);
#pragma unroll
    for (int j = 0; j < 4; j++)
      bf[j] = load_frag_b(&Bs[buf][(wn + j * 16) * LD], LD);
#pragma unroll
    for (int i = 0; i < 4; i++)
#pragma unroll
      for (int j = 0; j < 4; j++)
        acc[i][j] = wmma_f16(af[i], bf[j], acc[i][j]);

#if HAVE_TDM
    if (has_next && wave < 2) TDM_WAIT0();
#else
    if (has_next) {
      stage_tile_manual(Abase, As[buf ^ 1], BM, K, k0 + BK);
      stage_tile_manual(Bbase, Bs[buf ^ 1], BN, K, k0 + BK);
    }
#endif
    __syncthreads();
    buf ^= 1;
  }

  // ---- store: C/D layout -> global ----------------------------------------
  const int r0 = (lane >> 4) * 8;
  const int cN = lane & 15;
#pragma unroll
  for (int i = 0; i < 4; i++)
#pragma unroll
    for (int j = 0; j < 4; j++)
#pragma unroll
      for (int r = 0; r < 8; r++) {
        const size_t row = (size_t)(bm + wm + i * 16 + r0 + r);
        const size_t col = (size_t)(bn + wn + j * 16 + cN);
        if (OUT_F32)
          ((float*)Cout)[row * N + col] = acc[i][j][r];
        else
          ((_Float16*)Cout)[row * N + col] = (_Float16)acc[i][j][r];
      }
}

// ---------------------------------------------------------------------------
// Kernel 3: causal flash attention.
// Grid (SEQ/128, NUM_HEADS, BATCH), 256 threads = 8 waves.
// Wave w owns 16 query rows; Q in registers (scale folded). Key tiles of 64
// staged in LDS: K via TDM (row-padded), V manually transposed to [d][kk].
// ---------------------------------------------------------------------------
__global__ __launch_bounds__(256) void attn_kernel(
    const _Float16* __restrict__ qkv, _Float16* __restrict__ oh) {
  constexpr int KT  = 64;
  constexpr int LDK = HEAD_DIM + 8;  // 136 f16 (272B row, 16B aligned)
  constexpr int LDV = KT + 8;        // 72 f16 (144B row, 16B aligned)
  __shared__ _Float16 Ks [KT * LDK];        // K tile  [kk][d]
  __shared__ _Float16 Vts[HEAD_DIM * LDV];  // V tile transposed [d][kk]
  __shared__ _Float16 Pls[8 * 16 * LDV];    // per-wave P relayout buffers

  const int tid  = threadIdx.x;
  const int wave = tid >> 5;
  const int lane = tid & 31;
  const int head = blockIdx.y;
  const int b    = blockIdx.z;
  const int q0   = blockIdx.x * 128 + wave * 16;
  const int r0   = (lane >> 4) * 8;
  const int cN   = lane & 15;
  const size_t rowbase = (size_t)b * SEQ;

  // --- Q fragments once, softmax scale folded in ---------------------------
  v16h qf[4];
  {
    const _Float16* qp = qkv + (rowbase + q0) * H3 + head * HEAD_DIM;
#pragma unroll
    for (int d = 0; d < 4; d++) {
      v16h t = load_frag_a(qp + d * 32, H3);
      qf[d] = t * (_Float16)0.0883883476f;  // 1/sqrt(128)
    }
  }

  v8f   o[8] = {};
  float m_run[8], l_run[8];
#pragma unroll
  for (int r = 0; r < 8; r++) { m_run[r] = -3.0e38f; l_run[r] = 0.0f; }

  const int lr = tid >> 2;         // key row 0..63
  const int lc = (tid & 3) * 32;   // 32-d chunk
  const int nkt = 2 * blockIdx.x + 2;

  for (int kt = 0; kt < nkt; kt++) {
    const size_t krow0 = rowbase + (size_t)kt * KT;
    // ---- stage K tile (TDM) and transposed V tile (manual) ----------------
#if HAVE_TDM
    if (wave == 0)
      tdm_load_tile_f16(Ks, qkv + krow0 * H3 + HIDDEN + head * HEAD_DIM,
                        HEAD_DIM, KT, H3, /*256B rows*/ 5, /*16B pad*/ 3);
#else
    {
      const _Float16* kg = qkv + (krow0 + lr) * H3 + HIDDEN + head * HEAD_DIM + lc;
      _Float16* kd = Ks + lr * LDK + lc;
      *(v8h*)(kd)      = *(const v8h*)(kg);
      *(v8h*)(kd + 8)  = *(const v8h*)(kg + 8);
      *(v8h*)(kd + 16) = *(const v8h*)(kg + 16);
      *(v8h*)(kd + 24) = *(const v8h*)(kg + 24);
    }
#endif
    {
      const _Float16* vg = qkv + (krow0 + lr) * H3 + 2 * HIDDEN + head * HEAD_DIM + lc;
      v8h v0v = *(const v8h*)(vg);
      v8h v1v = *(const v8h*)(vg + 8);
      v8h v2v = *(const v8h*)(vg + 16);
      v8h v3v = *(const v8h*)(vg + 24);
#pragma unroll
      for (int i = 0; i < 8; i++) {
        Vts[(lc + i)      * LDV + lr] = v0v[i];
        Vts[(lc + 8 + i)  * LDV + lr] = v1v[i];
        Vts[(lc + 16 + i) * LDV + lr] = v2v[i];
        Vts[(lc + 24 + i) * LDV + lr] = v3v[i];
      }
    }
#if HAVE_TDM
    if (wave == 0) TDM_WAIT0();
#endif
    __syncthreads();

    if (kt * KT <= q0 + 15) {  // wave-uniform causal activity test
      // ---- S(16x64) = Q @ K^T : 16 WMMAs ----------------------------------
      v8f s[4] = {};
#pragma unroll
      for (int nt = 0; nt < 4; nt++)
#pragma unroll
        for (int d = 0; d < 4; d++)
          s[nt] = wmma_f16(qf[d],
                           load_frag_b(Ks + (nt * 16) * LDK + d * 32, LDK),
                           s[nt]);

      // ---- causal mask + online softmax -----------------------------------
#pragma unroll
      for (int r = 0; r < 8; r++) {
        const int qg = q0 + r0 + r;
        float tmax = -3.0e38f;
#pragma unroll
        for (int nt = 0; nt < 4; nt++) {
          const int kgidx = kt * KT + nt * 16 + cN;
          float sv = (kgidx <= qg) ? (float)s[nt][r] : -3.0e38f;
          s[nt][r] = sv;
          tmax = fmaxf(tmax, sv);
        }
#pragma unroll
        for (int off = 1; off < 16; off <<= 1)
          tmax = fmaxf(tmax, __shfl_xor(tmax, off, 32));
        const float mnew  = fmaxf(m_run[r], tmax);
        const float alpha = __expf(m_run[r] - mnew);
        float lsum = 0.0f;
#pragma unroll
        for (int nt = 0; nt < 4; nt++) {
          const float pv = __expf((float)s[nt][r] - mnew);
          s[nt][r] = pv;
          lsum += pv;
        }
#pragma unroll
        for (int off = 1; off < 16; off <<= 1)
          lsum += __shfl_xor(lsum, off, 32);
        l_run[r] = l_run[r] * alpha + lsum;
        m_run[r] = mnew;
#pragma unroll
        for (int j = 0; j < 8; j++) o[j][r] *= alpha;
      }

      // ---- P: D-layout -> A-layout via per-wave LDS buffer ----------------
      _Float16* Pl = Pls + wave * 16 * LDV;
#pragma unroll
      for (int nt = 0; nt < 4; nt++)
#pragma unroll
        for (int r = 0; r < 8; r++)
          Pl[(r0 + r) * LDV + nt * 16 + cN] = (_Float16)s[nt][r];
      __builtin_amdgcn_wave_barrier();  // LDS ops in-order within a wave

      // ---- O += P(16x64) @ V(64x128) : 16 WMMAs ---------------------------
#pragma unroll
      for (int kc = 0; kc < 2; kc++) {
        v16h pf = load_frag_a(Pl + kc * 32, LDV);
#pragma unroll
        for (int j = 0; j < 8; j++)
          o[j] = wmma_f16(pf,
                          load_frag_b(Vts + (j * 16) * LDV + kc * 32, LDV),
                          o[j]);
      }
    }
    __syncthreads();
  }

  // ---- epilogue: normalize, write merged-head layout [B*S, HIDDEN] --------
#pragma unroll
  for (int r = 0; r < 8; r++) {
    const float inv = 1.0f / l_run[r];
    const size_t orow = (rowbase + q0 + r0 + r) * HIDDEN + head * HEAD_DIM;
#pragma unroll
    for (int j = 0; j < 8; j++)
      oh[orow + j * 16 + cN] = (_Float16)(o[j][r] * inv);
  }
}

// ---------------------------------------------------------------------------
// Host launcher. Workspace (bytes):
//   [0,32M) x_f16 | [32M,56M) w_qkv_f16 | [56M,64M) w_out_f16
//   [64M,160M) qkv_f16 | [160M,192M) attn_out_f16      (192 MiB total)
// ---------------------------------------------------------------------------
extern "C" void kernel_launch(void* const* d_in, const int* in_sizes, int n_in,
                              void* d_out, int out_size, void* d_ws,
                              size_t ws_size, hipStream_t stream) {
  const float* x     = (const float*)d_in[0];
  const float* w_qkv = (const float*)d_in[1];
  const float* w_out = (const float*)d_in[2];
  float* out = (float*)d_out;

  char* ws = (char*)d_ws;
  _Float16* xh    = (_Float16*)(ws);
  _Float16* wqkvh = (_Float16*)(ws + (size_t)33554432);
  _Float16* wouth = (_Float16*)(ws + (size_t)58720256);
  _Float16* qkvh  = (_Float16*)(ws + (size_t)67108864);
  _Float16* ohbuf = (_Float16*)(ws + (size_t)167772160);

  const long nx = (long)MROWS * HIDDEN;   // 16,777,216
  const long nq = (long)H3 * HIDDEN;      // 12,582,912
  const long nw = (long)HIDDEN * HIDDEN;  //  4,194,304

  cvt_f32_to_f16<<<(nx / 8 + 255) / 256, 256, 0, stream>>>(x, xh, nx);
  cvt_f32_to_f16<<<(nq / 8 + 255) / 256, 256, 0, stream>>>(w_qkv, wqkvh, nq);
  cvt_f32_to_f16<<<(nw / 8 + 255) / 256, 256, 0, stream>>>(w_out, wouth, nw);

  // QKV projection: [8192,2048] x [6144,2048]^T -> f16 [8192,6144]
  gemm_wmma_kernel<false><<<dim3(H3 / 256, MROWS / 128), 256, 0, stream>>>(
      xh, wqkvh, (void*)qkvh, MROWS, H3, HIDDEN);

  // Causal MHA -> f16 [8192,2048] (heads merged)
  attn_kernel<<<dim3(SEQ / 128, NUM_HEADS, BATCH), 256, 0, stream>>>(qkvh,
                                                                     ohbuf);

  // Output projection: [8192,2048] x [2048,2048]^T -> f32 d_out
  gemm_wmma_kernel<true><<<dim3(HIDDEN / 256, MROWS / 128), 256, 0, stream>>>(
      ohbuf, wouth, (void*)out, MROWS, HIDDEN, HIDDEN);
}
